// PointNetHybridSeg_89438398972575
// MI455X (gfx1250) — matrix operand order repaired and checked by reading, not compiled, not run on hardware
//
#include <hip/hip_runtime.h>
#include <hip/hip_bf16.h>
#include <math.h>

// ---------------------------------------------------------------------------
// CDNA5 (gfx1250) implementation. wave32; all GEMM-like work goes through
// v_wmma_f32_16x16x32_f16 (f16 in, f32 accumulate). Activations NHWC f16.
// B fragments staged through LDS by multi-wave workgroups (async-to-LDS when
// the toolchain exposes the gfx1250 builtins).
// ---------------------------------------------------------------------------

typedef _Float16 v16h __attribute__((ext_vector_type(16)));
typedef float    v8f  __attribute__((ext_vector_type(8)));
typedef _Float16 h8   __attribute__((ext_vector_type(8)));
typedef int      v4i  __attribute__((ext_vector_type(4)));

#define F_RELU 1
#define F_UP   2
#define F_1X1  4

#if defined(__has_builtin)
#if __has_builtin(__builtin_amdgcn_global_load_async_to_lds_b128) && \
    __has_builtin(__builtin_amdgcn_s_wait_asynccnt)
#define HAS_ASYNC_LDS 1
#endif
#endif
#ifndef HAS_ASYNC_LDS
#define HAS_ASYNC_LDS 0
#endif

static __device__ __forceinline__ int iclampi(int v, int lo, int hi) {
  return v < lo ? lo : (v > hi ? hi : v);
}

// order-preserving float<->uint encoding for atomic max
static __device__ __forceinline__ unsigned encf(float f) {
  unsigned u = __float_as_uint(f);
  return (u & 0x80000000u) ? ~u : (u | 0x80000000u);
}
static __device__ __forceinline__ float decf(unsigned u) {
  unsigned v = (u & 0x80000000u) ? (u & 0x7fffffffu) : ~u;
  return __uint_as_float(v);
}
#define ENC_NEG_INF 0x007fffffu  // encf(-inf)

// Stage one 1KB B fragment (512 halfs) from global into LDS, cooperatively
// across the block. Uses gfx1250 async global->LDS DMA when available.
static __device__ __forceinline__ void stage_frag(const _Float16* __restrict__ src,
                                                  _Float16* shB, int tid, int nthreads) {
  for (int u = tid; u < 64; u += nthreads) {
#if HAS_ASYNC_LDS
    __builtin_amdgcn_global_load_async_to_lds_b128((v4i*)(src + u * 8), (v4i*)(shB + u * 8), 0,
                                                   0);
#else
    ((h8*)shB)[u] = ((const h8*)src)[u];
#endif
  }
#if HAS_ASYNC_LDS
  __builtin_amdgcn_s_wait_asynccnt(0);
#endif
}

// ---------------------------------------------------------------------------
// Weight packing into WMMA B-fragment layout.
// Fragment = 512 halfs, [lane*16 + e]; lane holds column n = nt*16 + lane%16,
// K index = kc*32 + (lane/16)*16 + e.
// ---------------------------------------------------------------------------
__global__ void pack_gemm_B_kernel(const float* __restrict__ W, int Kreal, int Nreal,
                                   int Kpad, int Npad, _Float16* __restrict__ Bf) {
  const int lane = threadIdx.x;
  const int NT = Npad >> 4;
  const int fi = blockIdx.x;
  const int kc = fi / NT, nt = fi % NT;
  const int n  = nt * 16 + (lane & 15);
  const int kb = kc * 32 + (lane >> 4) * 16;
  _Float16* o = Bf + (size_t)fi * 512 + lane * 16;
#pragma unroll
  for (int e = 0; e < 16; ++e) {
    const int k = kb + e;
    float v = (k < Kreal && n < Nreal) ? W[(size_t)k * Nreal + n] : 0.f;
    o[e] = (_Float16)v;
  }
}

// conv weights OIHW (Cout,Cin,ks,ks); frag index = (t*KC + kc)*NT + nt
__global__ void pack_conv_B_kernel(const float* __restrict__ W, int Cout, int Cin, int ksz,
                                   _Float16* __restrict__ Bf) {
  const int lane = threadIdx.x;
  const int NT = Cout >> 4, KC = Cin >> 5;
  const int fi = blockIdx.x;
  const int nt = fi % NT;
  const int kc = (fi / NT) % KC;
  const int t  = fi / (NT * KC);
  const int n  = nt * 16 + (lane & 15);
  const int kb = kc * 32 + (lane >> 4) * 16;
  _Float16* o = Bf + (size_t)fi * 512 + lane * 16;
#pragma unroll
  for (int e = 0; e < 16; ++e) {
    const int k = kb + e;
    o[e] = (_Float16)W[((size_t)n * Cin + k) * ksz + t];
  }
}

// y = acc*alpha + beta, alpha = g*rsqrt(1+eps), beta = b*alpha + be
__global__ void prep_affine_kernel(const float* g, const float* be, const float* b,
                                   float* alpha, float* beta, int nreal) {
  const int i = threadIdx.x;
  float a = 1.f, bt = 0.f;
  if (i < nreal) {
    a  = g ? g[i] * rsqrtf(1.f + 1e-5f) : 1.f;
    bt = (b ? b[i] : 0.f) * a + (be ? be[i] : 0.f);
  }
  alpha[i] = a;
  beta[i]  = bt;
}

// ---------------------------------------------------------------------------
// Generic dense GEMM: A (M x K) f16 row-major, Bf pre-packed fragments.
// Block = 4 waves; each wave does one 16x16 C tile (4 consecutive M tiles),
// all share the B fragment staged in LDS. grid = (N/16, M/64), block = 128.
// ---------------------------------------------------------------------------
__global__ void gemm_wmma_kernel(const _Float16* __restrict__ A, const _Float16* __restrict__ Bf,
                                 const float* __restrict__ alpha, const float* __restrict__ beta,
                                 _Float16* __restrict__ outH, float* __restrict__ outF,
                                 int N, int K, int flags) {
  __shared__ v16h shBv[32];
  _Float16* shB = (_Float16*)shBv;
  const int tid = threadIdx.x;
  const int wave = tid >> 5, lane = tid & 31;
  const int hh = lane >> 4, ln = lane & 15;
  const int nt = blockIdx.x;
  const int mt = blockIdx.y * 4 + wave;
  const int NT = N >> 4, KC = K >> 5;
  const size_t row = (size_t)mt * 16 + ln;
  const _Float16* ar = A + row * K;
  v8f acc = {};
  for (int kc = 0; kc < KC; ++kc) {
    stage_frag(Bf + ((size_t)(kc * NT + nt)) * 512, shB, tid, blockDim.x);
    const int kb = kc * 32 + hh * 16;
    if (kc + 1 < KC) __builtin_prefetch(ar + kb + 32, 0, 1);
    h8 lo = *(const h8*)(ar + kb);
    h8 hi = *(const h8*)(ar + kb + 8);
    v16h a;
#pragma unroll
    for (int e = 0; e < 8; ++e) { a[e] = lo[e]; a[8 + e] = hi[e]; }
    __syncthreads();
    const v16h bf = *(const v16h*)(shB + lane * 16);
    acc = __builtin_amdgcn_wmma_f32_16x16x32_f16(false, a, false, bf, (short)0, acc, false, false);
    __syncthreads();
  }
  const int n = nt * 16 + ln;
  const float al = alpha[n], bt = beta[n];
#pragma unroll
  for (int v = 0; v < 8; ++v) {
    float y = acc[v] * al + bt;
    if (flags & F_RELU) y = fmaxf(y, 0.f);
    const size_t m = (size_t)mt * 16 + v + 8 * hh;
    if (outH) outH[m * N + n] = (_Float16)y;
    if (outF) outF[m * N + n] = y;
  }
}

// ---------------------------------------------------------------------------
// Implicit-GEMM 3x3 (or 1x1) conv over NHWC f16, SAME padding.
// Optional 2x nearest upsampling of the (single) source, optional channel
// concat of two sources. Block = WX waves over x-tiles sharing the B fragment
// via LDS; one wave -> 16 pixels x 16 output channels.
// grid = ((W/16/WX)*(Cout/16), H, B), block = 32*WX.
// ---------------------------------------------------------------------------
__global__ void conv_wmma_kernel(const _Float16* __restrict__ s1, int C1,
                                 const _Float16* __restrict__ s2, int C2,
                                 const _Float16* __restrict__ Bf,
                                 const float* __restrict__ alpha, const float* __restrict__ beta,
                                 _Float16* __restrict__ out,
                                 int H, int W, int Cin, int Cout, int flags) {
  __builtin_amdgcn_s_wait_tensorcnt(0);
  __shared__ v16h shBv[32];
  _Float16* shB = (_Float16*)shBv;
  const int tid = threadIdx.x;
  const int wave = tid >> 5, lane = tid & 31;
  const int hh = lane >> 4, ln = lane & 15;
  const int XT = W >> 4, NT = Cout >> 4;
  const int WXV = (int)(blockDim.x >> 5);
  const int XB = XT / WXV;
  const int nt = blockIdx.x / XB;
  const int xt = (blockIdx.x % XB) * WXV + wave;
  const int y = blockIdx.y, b = blockIdx.z;
  const int KC = Cin >> 5;
  const bool k1 = (flags & F_1X1) != 0;
  const bool up = (flags & F_UP) != 0;
  const int taps = k1 ? 1 : 9;
  const int sH = up ? (H >> 1) : H, sW = up ? (W >> 1) : W;
  const int xm = xt * 16 + ln;  // pixel handled by this lane's A row
  v8f acc = {};
  for (int t = 0; t < taps; ++t) {
    const int dy = k1 ? 0 : (t / 3 - 1);
    const int dx = k1 ? 0 : (t % 3 - 1);
    const int iy = y + dy, ix = xm + dx;
    const bool valid = (iy >= 0) && (iy < H) && (ix >= 0) && (ix < W);
    const int sy = up ? (iy >> 1) : iy;
    const int sx = up ? (ix >> 1) : ix;
    for (int kc = 0; kc < KC; ++kc) {
      stage_frag(Bf + ((size_t)((t * KC + kc) * NT + nt)) * 512, shB, tid, blockDim.x);
      const int cb = kc * 32;
      const _Float16* sp;
      int cs, cof;
      if (s2 && cb >= C1) { sp = s2; cs = C2; cof = cb - C1; }
      else { sp = s1; cs = C1; cof = cb; }
      h8 lo = {}, hi = {};
      if (valid) {
        const _Float16* p = sp + (((size_t)(b * sH + sy) * sW + sx) * cs + cof + hh * 16);
        lo = *(const h8*)p;
        hi = *(const h8*)(p + 8);
      }
      v16h a;
#pragma unroll
      for (int e = 0; e < 8; ++e) { a[e] = lo[e]; a[8 + e] = hi[e]; }
      __syncthreads();
      const v16h bf = *(const v16h*)(shB + lane * 16);
      acc = __builtin_amdgcn_wmma_f32_16x16x32_f16(false, a, false, bf, (short)0, acc, false, false);
      __syncthreads();
    }
  }
  const int n = nt * 16 + ln;
  const float al = alpha[n], bt = beta[n];
#pragma unroll
  for (int v = 0; v < 8; ++v) {
    float yv = acc[v] * al + bt;
    if (flags & F_RELU) yv = fmaxf(yv, 0.f);
    const int xo = xt * 16 + v + 8 * hh;
    out[((size_t)(b * H + y) * W + xo) * Cout + n] = (_Float16)yv;
  }
}

// ---------------------------------------------------------------------------
// Elementwise / gather / scatter kernels
// ---------------------------------------------------------------------------
__global__ void build_msg_kernel(const float* __restrict__ x, const float* __restrict__ pos,
                                 const int* __restrict__ batch, _Float16* __restrict__ msg,
                                 int Npts) {
  const int r = blockIdx.x * blockDim.x + threadIdx.x;
  if (r >= Npts * 16) return;
  const int i = r >> 4, k = r & 15;
  const int nb = batch[i] * 2048 + k;
  _Float16* o = msg + (size_t)r * 32;
  o[0] = (_Float16)x[nb];
  o[1] = (_Float16)(pos[2 * nb + 0] - pos[2 * i + 0]);
  o[2] = (_Float16)(pos[2 * nb + 1] - pos[2 * i + 1]);
  for (int c = 3; c < 32; ++c) o[c] = (_Float16)0.f;
}

__global__ void init_pix_kernel(unsigned* __restrict__ pix, size_t n) {
  const size_t t = (size_t)blockIdx.x * blockDim.x + threadIdx.x;
  if (t < n) pix[t] = ENC_NEG_INF;
}

__global__ void scatter_max_kernel(const _Float16* __restrict__ featall,
                                   const float* __restrict__ pos, const int* __restrict__ batch,
                                   unsigned* __restrict__ pix, int Npts) {
  const int t = blockIdx.x * blockDim.x + threadIdx.x;
  if (t >= Npts * 128) return;
  const int i = t >> 7, c = t & 127;
  float m = -3.0e38f;
  for (int k = 0; k < 16; ++k)
    m = fmaxf(m, (float)featall[(size_t)(i * 16 + k) * 128 + c]);
  const int r  = iclampi((int)(pos[2 * i + 1] * 128.f), 0, 127);
  const int cl = iclampi((int)(pos[2 * i + 0] * 128.f), 0, 127);
  const size_t pidx = ((size_t)batch[i] * 128 + r) * 128 + cl;
  atomicMax(&pix[pidx * 128 + c], encf(m));
}

__global__ void decode_images_kernel(const unsigned* __restrict__ pix,
                                     _Float16* __restrict__ img, size_t n) {
  const size_t t = (size_t)blockIdx.x * blockDim.x + threadIdx.x;
  if (t >= n) return;
  const unsigned u = pix[t];
  float f = (u == ENC_NEG_INF) ? 0.f : decf(u);
  img[t] = (_Float16)f;
}

__global__ void pool_kernel(const _Float16* __restrict__ src, _Float16* __restrict__ dst,
                            int H, int W, int C) {  // input dims; output H/2 x W/2
  const int Ho = H >> 1, Wo = W >> 1;
  const int t = blockIdx.x * blockDim.x + threadIdx.x;
  const int total = 8 * Ho * Wo * C;
  if (t >= total) return;
  const int c = t % C;
  const int xw = (t / C) % Wo;
  const int yh = (t / (C * Wo)) % Ho;
  const int b = t / (C * Wo * Ho);
  const size_t base = ((size_t)(b * H + 2 * yh) * W + 2 * xw) * C + c;
  float m = (float)src[base];
  m = fmaxf(m, (float)src[base + C]);
  m = fmaxf(m, (float)src[base + (size_t)W * C]);
  m = fmaxf(m, (float)src[base + (size_t)W * C + C]);
  dst[((size_t)(b * Ho + yh) * Wo + xw) * C + c] = (_Float16)m;
}

__global__ void gather_x2_kernel(const _Float16* __restrict__ imf, const float* __restrict__ pos,
                                 const int* __restrict__ batch, _Float16* __restrict__ x2,
                                 int Npts) {
  const int t = blockIdx.x * blockDim.x + threadIdx.x;
  if (t >= Npts * 128) return;
  const int i = t >> 7, c = t & 127;
  const int r  = iclampi((int)(pos[2 * i + 1] * 128.f), 0, 127);
  const int cl = iclampi((int)(pos[2 * i + 0] * 128.f), 0, 127);
  x2[(size_t)i * 128 + c] =
      imf[(((size_t)batch[i] * 128 + r) * 128 + cl) * 128 + c];
}

__global__ void knn_kernel(const float* __restrict__ pos, int* __restrict__ nni,
                           float* __restrict__ nnd, int Npts) {
  const int i = blockIdx.x * blockDim.x + threadIdx.x;
  if (i >= Npts) return;
  const int jb = (i >> 11) << 11;  // batch start
  const float px = pos[2 * i], py = pos[2 * i + 1];
  float bd[8];
  int bi[8];
#pragma unroll
  for (int k = 0; k < 8; ++k) { bd[k] = 3.0e38f; bi[k] = jb; }
  for (int j = 0; j < 2048; ++j) {
    const float dx = pos[2 * (jb + j)] - px;
    const float dy = pos[2 * (jb + j) + 1] - py;
    float d = dx * dx + dy * dy;
    int jj = jb + j;
#pragma unroll
    for (int k = 0; k < 8; ++k) {
      if (d < bd[k]) {
        float td = bd[k]; bd[k] = d; d = td;
        int ti = bi[k]; bi[k] = jj; jj = ti;
      }
    }
  }
#pragma unroll
  for (int k = 0; k < 8; ++k) { nni[i * 8 + k] = bi[k]; nnd[i * 8 + k] = bd[k]; }
}

// fpin rows of 160 halfs: [0..127] = interp, [128] = x, [129..159] = 0
__global__ void build_fpin_kernel(const _Float16* __restrict__ x2, const float* __restrict__ xin,
                                  const int* __restrict__ nni, const float* __restrict__ nnd,
                                  _Float16* __restrict__ fpin, int Npts) {
  const int t = blockIdx.x * blockDim.x + threadIdx.x;
  if (t >= Npts * 160) return;
  const int i = t / 160, c = t % 160;
  float v;
  if (c < 128) {
    float ws = 0.f, s = 0.f;
#pragma unroll
    for (int k = 0; k < 8; ++k) {
      const float w = 1.f / fmaxf(nnd[i * 8 + k], 1e-16f);
      ws += w;
      s += w * (float)x2[(size_t)nni[i * 8 + k] * 128 + c];
    }
    v = s / ws;
  } else if (c == 128) {
    v = xin[i];
  } else {
    v = 0.f;
  }
  fpin[(size_t)i * 160 + c] = (_Float16)v;
}

__global__ void logsoftmax_kernel(const float* __restrict__ hf, float* __restrict__ out,
                                  int Npts) {
  const int i = blockIdx.x * blockDim.x + threadIdx.x;
  if (i >= Npts) return;
  float v[8], m = -3.0e38f;
#pragma unroll
  for (int c = 0; c < 8; ++c) { v[c] = hf[(size_t)i * 16 + c]; m = fmaxf(m, v[c]); }
  float s = 0.f;
#pragma unroll
  for (int c = 0; c < 8; ++c) s += expf(v[c] - m);
  const float ls = logf(s);
#pragma unroll
  for (int c = 0; c < 8; ++c) out[(size_t)i * 8 + c] = v[c] - m - ls;
}

// ---------------------------------------------------------------------------
// Host orchestration
// ---------------------------------------------------------------------------

// Input index map: top-level dict order (x, pos, params, batch); params
// flattened in JAX pytree (recursively sorted keys) order.
enum {
  IX_X = 0, IX_POS = 1,
  FP_W1 = 2, FP_W2, FP_W3, FP_B1, FP_B2, FP_B3, FP_BE1, FP_BE2, FP_G1, FP_G2,
  HD_W1 = 12, HD_W2, HD_W3, HD_B1, HD_B2, HD_B3, HD_BE1, HD_BE2, HD_G1, HD_G2,
  SA_W1 = 22, SA_W2, SA_W3, SA_B1, SA_B2, SA_B3, SA_BE1, SA_BE2, SA_G1, SA_G2,
  UN_BOUT = 32,
  C1_B1 = 33, C1_B2, C1_BE1, C1_BE2, C1_G1, C1_G2, C1_W1, C1_W2,
  C2_B1 = 41, C2_B2, C2_BE1, C2_BE2, C2_G1, C2_G2, C2_W1, C2_W2,
  C3_B1 = 49, C3_B2, C3_BE1, C3_BE2, C3_G1, C3_G2, C3_W1, C3_W2,
  U2_B = 57, U2_BE, U2_G, U2_W,
  U3_B = 61, U3_BE, U3_G, U3_W,
  UC2_B1 = 65, UC2_B2, UC2_BE1, UC2_BE2, UC2_G1, UC2_G2, UC2_W1, UC2_W2,
  UC3_B1 = 73, UC3_B2, UC3_BE1, UC3_BE2, UC3_G1, UC3_G2, UC3_W1, UC3_W2,
  UN_WOUT = 81,
  IX_BATCH = 82,
  N_IN_EXPECTED = 83
};

extern "C" void kernel_launch(void* const* d_in, const int* in_sizes, int n_in,
                              void* d_out, int out_size, void* d_ws, size_t ws_size,
                              hipStream_t stream) {
  (void)in_sizes; (void)out_size; (void)ws_size;
  if (n_in < N_IN_EXPECTED) return;

  const int Npts = 16384;  // B*P
  const float* x_in  = (const float*)d_in[IX_X];
  const float* pos   = (const float*)d_in[IX_POS];
  const int*   batch = (const int*)d_in[IX_BATCH];
  float* out = (float*)d_out;
  auto F = [&](int i) { return (const float*)d_in[i]; };

  // bump allocator on workspace
  char* base = (char*)d_ws;
  size_t off = 0;
  auto alloc = [&](size_t bytes) -> void* {
    size_t o = (off + 255) & ~(size_t)255;
    off = o + bytes;
    return (void*)(base + o);
  };
  auto allocH = [&](size_t nh) { return (_Float16*)alloc(nh * sizeof(_Float16)); };
  auto allocF = [&](size_t nf) { return (float*)alloc(nf * sizeof(float)); };

  auto nb256 = [](size_t total) { return (unsigned)((total + 255) / 256); };

  // ---- pack weights into B fragments
  auto packG = [&](const float* W, int Kreal, int Nreal, int Kpad, int Npad) -> const _Float16* {
    const int nf = (Kpad / 32) * (Npad / 16);
    _Float16* p = allocH((size_t)nf * 512);
    pack_gemm_B_kernel<<<nf, 32, 0, stream>>>(W, Kreal, Nreal, Kpad, Npad, p);
    return p;
  };
  auto packC = [&](const float* W, int Cout, int Cin, int ksz) -> const _Float16* {
    const int nf = ksz * (Cin / 32) * (Cout / 16);
    _Float16* p = allocH((size_t)nf * 512);
    pack_conv_B_kernel<<<nf, 32, 0, stream>>>(W, Cout, Cin, ksz, p);
    return p;
  };
  auto affine = [&](const float* g, const float* be, const float* b, int nreal,
                    int npad) -> const float* {
    float* p = allocF(2 * (size_t)npad);
    prep_affine_kernel<<<1, npad, 0, stream>>>(g, be, b, p, p + npad, nreal);
    return p;
  };

  // MLP weights
  const _Float16* saB1 = packG(F(SA_W1), 3, 64, 32, 64);
  const _Float16* saB2 = packG(F(SA_W2), 64, 64, 64, 64);
  const _Float16* saB3 = packG(F(SA_W3), 64, 128, 64, 128);
  const _Float16* fpB1 = packG(F(FP_W1), 129, 128, 160, 128);
  const _Float16* fpB2 = packG(F(FP_W2), 128, 128, 128, 128);
  const _Float16* fpB3 = packG(F(FP_W3), 128, 128, 128, 128);
  const _Float16* hdB1 = packG(F(HD_W1), 128, 128, 128, 128);
  const _Float16* hdB2 = packG(F(HD_W2), 128, 128, 128, 128);
  const _Float16* hdB3 = packG(F(HD_W3), 128, 8, 128, 16);
  // conv weights
  const _Float16* c1B1  = packC(F(C1_W1), 64, 128, 9);
  const _Float16* c1B2  = packC(F(C1_W2), 64, 64, 9);
  const _Float16* c2B1  = packC(F(C2_W1), 128, 64, 9);
  const _Float16* c2B2  = packC(F(C2_W2), 128, 128, 9);
  const _Float16* c3B1  = packC(F(C3_W1), 256, 128, 9);
  const _Float16* c3B2  = packC(F(C3_W2), 256, 256, 9);
  const _Float16* u3B   = packC(F(U3_W), 128, 256, 9);
  const _Float16* uc3B1 = packC(F(UC3_W1), 128, 256, 9);
  const _Float16* uc3B2 = packC(F(UC3_W2), 128, 128, 9);
  const _Float16* u2B   = packC(F(U2_W), 64, 128, 9);
  const _Float16* uc2B1 = packC(F(UC2_W1), 64, 128, 9);
  const _Float16* uc2B2 = packC(F(UC2_W2), 64, 64, 9);
  const _Float16* woutB = packC(F(UN_WOUT), 128, 64, 1);

  // affine (alpha/beta) params
  const float* afSA1 = affine(F(SA_G1), F(SA_BE1), F(SA_B1), 64, 64);
  const float* afSA2 = affine(F(SA_G2), F(SA_BE2), F(SA_B2), 64, 64);
  const float* afSA3 = affine(nullptr, nullptr, F(SA_B3), 128, 128);
  const float* afFP1 = affine(F(FP_G1), F(FP_BE1), F(FP_B1), 128, 128);
  const float* afFP2 = affine(F(FP_G2), F(FP_BE2), F(FP_B2), 128, 128);
  const float* afFP3 = affine(nullptr, nullptr, F(FP_B3), 128, 128);
  const float* afHD1 = affine(F(HD_G1), F(HD_BE1), F(HD_B1), 128, 128);
  const float* afHD2 = affine(F(HD_G2), F(HD_BE2), F(HD_B2), 128, 128);
  const float* afHD3 = affine(nullptr, nullptr, F(HD_B3), 8, 16);
  const float* afC1a = affine(F(C1_G1), F(C1_BE1), F(C1_B1), 64, 64);
  const float* afC1b = affine(F(C1_G2), F(C1_BE2), F(C1_B2), 64, 64);
  const float* afC2a = affine(F(C2_G1), F(C2_BE1), F(C2_B1), 128, 128);
  const float* afC2b = affine(F(C2_G2), F(C2_BE2), F(C2_B2), 128, 128);
  const float* afC3a = affine(F(C3_G1), F(C3_BE1), F(C3_B1), 256, 256);
  const float* afC3b = affine(F(C3_G2), F(C3_BE2), F(C3_B2), 256, 256);
  const float* afU3  = affine(F(U3_G), F(U3_BE), F(U3_B), 128, 128);
  const float* afD3a = affine(F(UC3_G1), F(UC3_BE1), F(UC3_B1), 128, 128);
  const float* afD3b = affine(F(UC3_G2), F(UC3_BE2), F(UC3_B2), 128, 128);
  const float* afU2  = affine(F(U2_G), F(U2_BE), F(U2_B), 64, 64);
  const float* afD2a = affine(F(UC2_G1), F(UC2_BE1), F(UC2_B1), 64, 64);
  const float* afD2b = affine(F(UC2_G2), F(UC2_BE2), F(UC2_B2), 64, 64);
  const float* afOut = affine(nullptr, nullptr, F(UN_BOUT), 128, 128);

  // ---- activation buffers
  const size_t R = (size_t)Npts * 16;                 // SA rows
  _Float16* msg     = allocH(R * 32);
  _Float16* saH1    = allocH(R * 64);
  _Float16* saH2    = allocH(R * 64);
  _Float16* featall = allocH(R * 128);
  unsigned* pixU    = (unsigned*)alloc((size_t)131072 * 128 * 4);
  _Float16* imgH    = allocH((size_t)131072 * 128);
  _Float16* e1a = allocH((size_t)8 * 128 * 128 * 64);
  _Float16* e1  = allocH((size_t)8 * 128 * 128 * 64);
  _Float16* e1p = allocH((size_t)8 * 64 * 64 * 64);
  _Float16* e2a = allocH((size_t)8 * 64 * 64 * 128);
  _Float16* e2  = allocH((size_t)8 * 64 * 64 * 128);
  _Float16* e2p = allocH((size_t)8 * 32 * 32 * 128);
  _Float16* e3a = allocH((size_t)8 * 32 * 32 * 256);
  _Float16* e3  = allocH((size_t)8 * 32 * 32 * 256);
  _Float16* up3 = allocH((size_t)8 * 64 * 64 * 128);
  _Float16* d3a = allocH((size_t)8 * 64 * 64 * 128);
  _Float16* d3  = allocH((size_t)8 * 64 * 64 * 128);
  _Float16* up2 = allocH((size_t)8 * 128 * 128 * 64);
  _Float16* d2a = allocH((size_t)8 * 128 * 128 * 64);
  _Float16* d2  = allocH((size_t)8 * 128 * 128 * 64);
  _Float16* imf = allocH((size_t)8 * 128 * 128 * 128);
  _Float16* x2  = allocH((size_t)Npts * 128);
  int*      nni = (int*)alloc((size_t)Npts * 8 * 4);
  float*    nnd = allocF((size_t)Npts * 8);
  _Float16* fpin = allocH((size_t)Npts * 160);
  _Float16* f1 = allocH((size_t)Npts * 128);
  _Float16* f2 = allocH((size_t)Npts * 128);
  _Float16* f3 = allocH((size_t)Npts * 128);
  _Float16* g1 = allocH((size_t)Npts * 128);
  _Float16* g2 = allocH((size_t)Npts * 128);
  float*    hf = allocF((size_t)Npts * 16);

  auto gemm = [&](const _Float16* A, const _Float16* Bf, const float* af, int M, int N, int K,
                  int flags, _Float16* oh, float* of) {
    dim3 grid(N / 16, M / 64);  // 4 waves/block over M tiles
    gemm_wmma_kernel<<<grid, 128, 0, stream>>>(A, Bf, af, af + N, oh, of, N, K, flags);
  };
  auto conv = [&](const _Float16* s1, int C1, const _Float16* s2, int C2, const _Float16* Bf,
                  const float* af, _Float16* o, int H, int W, int Cin, int Cout, int flags) {
    const int XT = W / 16;
    const int WX = XT >= 4 ? 4 : XT;
    dim3 grid((XT / WX) * (Cout / 16), H, 8);
    conv_wmma_kernel<<<grid, 32 * WX, 0, stream>>>(s1, C1, s2, C2, Bf, af, af + Cout, o, H, W,
                                                   Cin, Cout, flags);
  };

  // ---- Stage 1: SA MLP over neighbors + max
  build_msg_kernel<<<nb256(R), 256, 0, stream>>>(x_in, pos, batch, msg, Npts);
  gemm(msg, saB1, afSA1, (int)R, 64, 32, F_RELU, saH1, nullptr);
  gemm(saH1, saB2, afSA2, (int)R, 64, 64, F_RELU, saH2, nullptr);
  gemm(saH2, saB3, afSA3, (int)R, 128, 64, 0, featall, nullptr);

  // ---- Stage 2: scatter segment-max onto 128x128 grid
  const size_t pixN = (size_t)131072 * 128;
  init_pix_kernel<<<nb256(pixN), 256, 0, stream>>>(pixU, pixN);
  scatter_max_kernel<<<nb256((size_t)Npts * 128), 256, 0, stream>>>(featall, pos, batch, pixU,
                                                                    Npts);
  decode_images_kernel<<<nb256(pixN), 256, 0, stream>>>(pixU, imgH, pixN);

  // ---- Stage 3: UNet (NHWC f16, implicit-GEMM convs)
  conv(imgH, 128, nullptr, 0, c1B1, afC1a, e1a, 128, 128, 128, 64, F_RELU);
  conv(e1a, 64, nullptr, 0, c1B2, afC1b, e1, 128, 128, 64, 64, F_RELU);
  pool_kernel<<<nb256((size_t)8 * 64 * 64 * 64), 256, 0, stream>>>(e1, e1p, 128, 128, 64);
  conv(e1p, 64, nullptr, 0, c2B1, afC2a, e2a, 64, 64, 64, 128, F_RELU);
  conv(e2a, 128, nullptr, 0, c2B2, afC2b, e2, 64, 64, 128, 128, F_RELU);
  pool_kernel<<<nb256((size_t)8 * 32 * 32 * 128), 256, 0, stream>>>(e2, e2p, 64, 64, 128);
  conv(e2p, 128, nullptr, 0, c3B1, afC3a, e3a, 32, 32, 128, 256, F_RELU);
  conv(e3a, 256, nullptr, 0, c3B2, afC3b, e3, 32, 32, 256, 256, F_RELU);
  conv(e3, 256, nullptr, 0, u3B, afU3, up3, 64, 64, 256, 128, F_RELU | F_UP);
  conv(e2, 128, up3, 128, uc3B1, afD3a, d3a, 64, 64, 256, 128, F_RELU);
  conv(d3a, 128, nullptr, 0, uc3B2, afD3b, d3, 64, 64, 128, 128, F_RELU);
  conv(d3, 128, nullptr, 0, u2B, afU2, up2, 128, 128, 128, 64, F_RELU | F_UP);
  conv(e1, 64, up2, 64, uc2B1, afD2a, d2a, 128, 128, 128, 64, F_RELU);
  conv(d2a, 64, nullptr, 0, uc2B2, afD2b, d2, 128, 128, 64, 64, F_RELU);
  conv(d2, 64, nullptr, 0, woutB, afOut, imf, 128, 128, 64, 128, F_1X1);

  // ---- Stage 4: gather + kNN feature propagation
  gather_x2_kernel<<<nb256((size_t)Npts * 128), 256, 0, stream>>>(imf, pos, batch, x2, Npts);
  knn_kernel<<<(Npts + 127) / 128, 128, 0, stream>>>(pos, nni, nnd, Npts);
  build_fpin_kernel<<<nb256((size_t)Npts * 160), 256, 0, stream>>>(x2, x_in, nni, nnd, fpin,
                                                                   Npts);

  // ---- Stage 5: FP MLP + head + log-softmax
  gemm(fpin, fpB1, afFP1, Npts, 128, 160, F_RELU, f1, nullptr);
  gemm(f1, fpB2, afFP2, Npts, 128, 128, F_RELU, f2, nullptr);
  gemm(f2, fpB3, afFP3, Npts, 128, 128, 0, f3, nullptr);
  gemm(f3, hdB1, afHD1, Npts, 128, 128, F_RELU, g1, nullptr);
  gemm(g1, hdB2, afHD2, Npts, 128, 128, F_RELU, g2, nullptr);
  gemm(g2, hdB3, afHD3, Npts, 16, 128, 0, nullptr, hf);
  logsoftmax_kernel<<<nb256(Npts), 256, 0, stream>>>(hf, out, Npts);
}
